// MyLSTM_62019327754429
// MI455X (gfx1250) — matrix operand order repaired and checked by reading, not compiled, not run on hardware
//
#include <hip/hip_runtime.h>

#define B_   1024
#define T_   512
#define H_   64
#define G_   256
#define BT   16
#define HS   72     // h-state row stride in halves (64 + 8 pad; 144B rows keep 16B alignment)
#define GSR  264    // gates row stride in floats (256 + 8 pad)

typedef __attribute__((ext_vector_type(16))) _Float16 v16h;
typedef __attribute__((ext_vector_type(8)))  _Float16 v8h;
typedef __attribute__((ext_vector_type(4)))  _Float16 v4h;
typedef __attribute__((ext_vector_type(8)))  float    v8f;
typedef __attribute__((ext_vector_type(4)))  float    v4f;

// Fast activations: v_exp_f32 + v_rcp_f32 only (no IEEE divide expansion).
__device__ __forceinline__ float sigm(float x) {
    return __builtin_amdgcn_rcpf(1.0f + __expf(-x));
}
// tanh(x) = 1 - 2/(e^{2x}+1): saturates cleanly to +-1 without inf/inf NaN
__device__ __forceinline__ float tanhfast(float x) {
    float r = __builtin_amdgcn_rcpf(__expf(2.0f * x) + 1.0f);
    return __builtin_fmaf(-2.0f, r, 1.0f);
}

// A-fragment (16x32 f16, MxK) per ISA layout:
// lanes 0-15: M=lane, K={0..7} in VGPR0-3, K={16..23} in VGPR4-7
// lanes 16-31: M=lane-16, K={8..15} and {24..31}
// => per lane two contiguous 8-half (16B) chunks from the row-major h buffer.
__device__ __forceinline__ v16h load_afrag(const _Float16* hb, int m, int ks, int baseK) {
    const v8h* p0 = (const v8h*)(hb + m * HS + ks + baseK);
    const v8h* p1 = (const v8h*)(hb + m * HS + ks + 16 + baseK);
    v8h lo = *p0, hi = *p1;
    v16h a;
#pragma unroll
    for (int i = 0; i < 8; ++i) { a[i] = lo[i]; a[8 + i] = hi[i]; }
    return a;
}

// B-fragment (32x16 f16, KxN) for gates = h @ W^T, W is [256][64] row-major.
// lanes 0-15 hold K={0..7,16..23}, lanes 16-31 hold K={8..15,24..31}; lane%16 = N column.
// Loaded ONCE before the time loop; lives in VGPRs for all 512 steps.
__device__ __forceinline__ v16h load_bfrag(const float* __restrict__ W, int ntile, int kslice, int lane) {
    int n  = ntile * 16 + (lane & 15);
    int kb = kslice + ((lane & 16) ? 8 : 0);
    const float* p = W + n * H_ + kb;
    v16h b;
#pragma unroll
    for (int j = 0; j < 16; ++j) b[j] = (_Float16)p[j < 8 ? j : j + 8];
    return b;
}

#define WMMA(A, Bf, C) __builtin_amdgcn_wmma_f32_16x16x32_f16(false, (A), false, (Bf), (short)0, (C), false, false)

__global__ __launch_bounds__(256)
void lstm2_fused(const float* __restrict__ x,
                 const float* __restrict__ Wih0, const float* __restrict__ Whh0,
                 const float* __restrict__ bih0, const float* __restrict__ bhh0,
                 const float* __restrict__ Wih1, const float* __restrict__ Whh1,
                 const float* __restrict__ bih1, const float* __restrict__ bhh1,
                 const float* __restrict__ Wfc,  const float* __restrict__ bfc,
                 float* __restrict__ out)
{
    __shared__ _Float16 h0s[BT * HS];
    __shared__ _Float16 h1s[BT * HS];
    __shared__ float    gs0[BT * GSR];   // gates for layer 0, step t (pipelined)
    __shared__ float    gs1[BT * GSR];   // gates for layer 1, step t-1
    __shared__ float    xs[BT * T_];

    const int tid  = threadIdx.x;
    const int wv   = tid >> 5;
    const int lane = tid & 31;
    const int b0   = blockIdx.x * BT;

    // Stage this tile's x (contiguous 16*512 floats since I==1) into LDS once.
    for (int i = tid; i < BT * T_; i += 256) xs[i] = x[b0 * T_ + i];
    // Zero h-state and the layer-0 gate buffer (gates0(0) = 0 since h0(-1)=0).
    for (int i = tid; i < BT * HS; i += 256) { h0s[i] = (_Float16)0.0f; h1s[i] = (_Float16)0.0f; }
    for (int i = tid; i < BT * GSR; i += 256) gs0[i] = 0.0f;

    // Preload all B fragments (loop-invariant weights) into VGPRs: 12 x v16h = 96 VGPRs.
    v16h Bh0[2][2], Bi1[2][2], Bh1[2][2];
#pragma unroll
    for (int i = 0; i < 2; ++i) {
#pragma unroll
        for (int k = 0; k < 2; ++k) {
            Bh0[i][k] = load_bfrag(Whh0, wv * 2 + i, k * 32, lane);
            Bi1[i][k] = load_bfrag(Wih1, wv * 2 + i, k * 32, lane);
            Bh1[i][k] = load_bfrag(Whh1, wv * 2 + i, k * 32, lane);
        }
    }

    // Per-thread elementwise constants (I==1 so layer-0 input term is scalar per gate).
    const int eb = tid >> 4;         // batch row 0..15
    const int eh = (tid & 15) << 2;  // h base: 0,4,...,60
    float wi0[4], wf0[4], wg0[4], wo0[4];
    float bi0[4], bf0[4], bg0[4], bo0[4];
    float bi1[4], bf1[4], bg1[4], bo1[4];
#pragma unroll
    for (int j = 0; j < 4; ++j) {
        int cI = eh + j, cF = cI + 64, cG = cI + 128, cO = cI + 192;
        wi0[j] = Wih0[cI]; wf0[j] = Wih0[cF]; wg0[j] = Wih0[cG]; wo0[j] = Wih0[cO];
        bi0[j] = bih0[cI] + bhh0[cI]; bf0[j] = bih0[cF] + bhh0[cF];
        bg0[j] = bih0[cG] + bhh0[cG]; bo0[j] = bih0[cO] + bhh0[cO];
        bi1[j] = bih1[cI] + bhh1[cI]; bf1[j] = bih1[cF] + bhh1[cF];
        bg1[j] = bih1[cG] + bhh1[cG]; bo1[j] = bih1[cO] + bhh1[cO];
    }
    float c0[4] = {0, 0, 0, 0}, c1[4] = {0, 0, 0, 0};

    const int m     = lane & 15;
    const int baseK = (lane & 16) ? 8 : 0;
    const int drow  = (lane & 16) ? 8 : 0;
    const int colA  = (wv * 2) * 16 + (lane & 15);
    const int colB  = (wv * 2 + 1) * 16 + (lane & 15);

    __syncthreads();

    // Software-pipelined: each iteration does EW-L0(t) + EW-L1(t-1), then a
    // merged WMMA burst producing gates1(t) and gates0(t+1). 2 barriers/step.
    for (int t = 0; t < T_; ++t) {
        // ---- elementwise phase ----
        {
            // layer 0, step t
            float xv = xs[eb * T_ + t];
            v4f gi = *(const v4f*)&gs0[eb * GSR + eh];
            v4f gf = *(const v4f*)&gs0[eb * GSR + eh + 64];
            v4f gg = *(const v4f*)&gs0[eb * GSR + eh + 128];
            v4f go = *(const v4f*)&gs0[eb * GSR + eh + 192];
            v4h hv;
#pragma unroll
            for (int j = 0; j < 4; ++j) {
                float ig = sigm(gi[j] + xv * wi0[j] + bi0[j]);
                float fg = sigm(gf[j] + xv * wf0[j] + bf0[j]);
                float gv = tanhfast(gg[j] + xv * wg0[j] + bg0[j]);
                float og = sigm(go[j] + xv * wo0[j] + bo0[j]);
                c0[j] = fg * c0[j] + ig * gv;
                hv[j] = (_Float16)(og * tanhfast(c0[j]));
            }
            *(v4h*)&h0s[eb * HS + eh] = hv;

            // layer 1, step t-1 (uniform branch; skipped only at t==0)
            if (t > 0) {
                v4f qi = *(const v4f*)&gs1[eb * GSR + eh];
                v4f qf = *(const v4f*)&gs1[eb * GSR + eh + 64];
                v4f qg = *(const v4f*)&gs1[eb * GSR + eh + 128];
                v4f qo = *(const v4f*)&gs1[eb * GSR + eh + 192];
                v4h hw;
#pragma unroll
                for (int j = 0; j < 4; ++j) {
                    float ig = sigm(qi[j] + bi1[j]);
                    float fg = sigm(qf[j] + bf1[j]);
                    float gv = tanhfast(qg[j] + bg1[j]);
                    float og = sigm(qo[j] + bo1[j]);
                    c1[j] = fg * c1[j] + ig * gv;
                    hw[j] = (_Float16)(og * tanhfast(c1[j]));
                }
                *(v4h*)&h1s[eb * HS + eh] = hw;
            }
        }
        __syncthreads();

        // ---- merged WMMA burst: gates1(t) = h1(t-1)@Whh1^T + h0(t)@Wih1^T,
        //                          gates0(t+1) = h0(t)@Whh0^T ----
        {
            v16h hp0 = load_afrag(h1s, m, 0, baseK);
            v16h hp1 = load_afrag(h1s, m, 32, baseK);
            v16h u0  = load_afrag(h0s, m, 0, baseK);
            v16h u1  = load_afrag(h0s, m, 32, baseK);

            v8f g1a = {}, g1b = {}, g0a = {}, g0b = {};
            // interleave 4 independent accumulator chains to hide WMMA RAW latency
            g1a = WMMA(hp0, Bh1[0][0], g1a);
            g1b = WMMA(hp0, Bh1[1][0], g1b);
            g0a = WMMA(u0,  Bh0[0][0], g0a);
            g0b = WMMA(u0,  Bh0[1][0], g0b);
            g1a = WMMA(hp1, Bh1[0][1], g1a);
            g1b = WMMA(hp1, Bh1[1][1], g1b);
            g0a = WMMA(u1,  Bh0[0][1], g0a);
            g0b = WMMA(u1,  Bh0[1][1], g0b);
            g1a = WMMA(u0,  Bi1[0][0], g1a);
            g1b = WMMA(u0,  Bi1[1][0], g1b);
            g1a = WMMA(u1,  Bi1[0][1], g1a);
            g1b = WMMA(u1,  Bi1[1][1], g1b);

#pragma unroll
            for (int r = 0; r < 8; ++r) {
                gs1[(drow + r) * GSR + colA] = g1a[r];
                gs1[(drow + r) * GSR + colB] = g1b[r];
                gs0[(drow + r) * GSR + colA] = g0a[r];
                gs0[(drow + r) * GSR + colB] = g0b[r];
            }
        }
        __syncthreads();
    }

    // ---- epilogue: elementwise layer 1 for step T-1 ----
    {
        v4f qi = *(const v4f*)&gs1[eb * GSR + eh];
        v4f qf = *(const v4f*)&gs1[eb * GSR + eh + 64];
        v4f qg = *(const v4f*)&gs1[eb * GSR + eh + 128];
        v4f qo = *(const v4f*)&gs1[eb * GSR + eh + 192];
        v4h hw;
#pragma unroll
        for (int j = 0; j < 4; ++j) {
            float ig = sigm(qi[j] + bi1[j]);
            float fg = sigm(qf[j] + bf1[j]);
            float gv = tanhfast(qg[j] + bg1[j]);
            float og = sigm(qo[j] + bo1[j]);
            c1[j] = fg * c1[j] + ig * gv;
            hw[j] = (_Float16)(og * tanhfast(c1[j]));
        }
        *(v4h*)&h1s[eb * HS + eh] = hw;
    }
    __syncthreads();

    // ---- final FC: out[b] = h1[T-1] . Wfc + bfc (O == 1) ----
    if (tid < BT) {
        float s = bfc[0];
#pragma unroll 8
        for (int hh = 0; hh < H_; ++hh) s += (float)h1s[tid * HS + hh] * Wfc[hh];
        out[b0 + tid] = s;
    }
}

extern "C" void kernel_launch(void* const* d_in, const int* in_sizes, int n_in,
                              void* d_out, int out_size, void* d_ws, size_t ws_size,
                              hipStream_t stream) {
    (void)in_sizes; (void)n_in; (void)out_size; (void)d_ws; (void)ws_size;
    const float* x    = (const float*)d_in[0];
    const float* Wih0 = (const float*)d_in[1];
    const float* Whh0 = (const float*)d_in[2];
    const float* bih0 = (const float*)d_in[3];
    const float* bhh0 = (const float*)d_in[4];
    const float* Wih1 = (const float*)d_in[5];
    const float* Whh1 = (const float*)d_in[6];
    const float* bih1 = (const float*)d_in[7];
    const float* bhh1 = (const float*)d_in[8];
    const float* Wfc  = (const float*)d_in[9];
    const float* bfc  = (const float*)d_in[10];
    float* out = (float*)d_out;

    hipLaunchKernelGGL(lstm2_fused, dim3(B_ / BT), dim3(256), 0, stream,
                       x, Wih0, Whh0, bih0, bhh0, Wih1, Whh1, bih1, bhh1, Wfc, bfc, out);
}
